// SingleForgetGateTreeLSTM_16063177687520
// MI455X (gfx1250) — matrix operand order repaired and checked by reading, not compiled, not run on hardware
//
#include <hip/hip_runtime.h>
#include <hip/hip_bf16.h>

// CDNA5 (gfx1250) TreeLSTM, fused per-level kernels using v_wmma_f32_16x16x32_bf16.
// Design: bandwidth-bound problem (~0.8 GB traffic vs 150 GFLOP), so fuse
// W-GEMM + U-GEMM + gates per tree level; never spill the 537MB wx to HBM.

typedef __attribute__((ext_vector_type(16))) __bf16 v16bf;
typedef __attribute__((ext_vector_type(8)))  float  v8f;

#define HS   256
#define GATES 1024      // 4*HS
#define NT_TOT 64       // 1024/16 column tiles
#define XK   300
#define XKP  320        // padded K for x (10 chunks of 32)
#define KC_W 10
#define KC_U 16         // 512/32

// ---------------------------------------------------------------------------
// Pack a (1024 x K) row-major fp32 weight into WMMA B-fragment order, bf16:
//   out[kc][nt][lane][e] = W[nt*16+lane%16, kc*32+kk(e,lane/16)]  (zero if k >= K)
// B layout (32x16, 16-bit): lane%16 indexes N within the tile; the 16 bf16
// per lane index K: e<8 -> 8h+e, e>=8 -> 16+8h+(e-8) -> each lane loads 32
// contiguous bytes at GEMM time (two global_load_b128).
// ---------------------------------------------------------------------------
__global__ void pack_b_kernel(const float* __restrict__ W, __bf16* __restrict__ out,
                              int K, int total)
{
    int idx = blockIdx.x * blockDim.x + threadIdx.x;
    if (idx >= total) return;
    int e    =  idx        & 15;
    int lane = (idx >> 4)  & 31;
    int nt   = (idx >> 9)  & 63;
    int kc   =  idx >> 15;
    int k = kc * 32 + ((e < 8) ? ((lane >> 4) * 8 + e) : (16 + (lane >> 4) * 8 + (e - 8)));
    int j = nt * 16 + (lane & 15);
    float v = (k < K) ? W[(size_t)j * K + k] : 0.0f;
    out[idx] = (__bf16)v;
}

// x (N x 300 f32) -> xbf (N x 320 bf16, zero-padded)
__global__ void conv_x_kernel(const float* __restrict__ x, __bf16* __restrict__ xbf,
                              long total)
{
    long idx = (long)blockIdx.x * blockDim.x + threadIdx.x;
    if (idx >= total) return;
    long row = idx / XKP;
    int  k   = (int)(idx - row * XKP);
    xbf[idx] = (k < XK) ? (__bf16)x[row * XK + k] : (__bf16)0.0f;
}

// ---------------------------------------------------------------------------
// One tree level: wx = x_l @ W^T (+ ch_h @ U^T) + b, then gates.
// Block = 256 threads (8 wave32), 16 nodes per block, 1024 columns split
// 8 x 16-col tiles per wave. A tiles staged in LDS (bf16); accumulators
// exchanged through LDS (reusing staging space) for the cross-gate phase.
// ---------------------------------------------------------------------------
__global__ __launch_bounds__(256)
void tree_level_kernel(const __bf16* __restrict__ xbf,
                       const __bf16* __restrict__ Wp,
                       const __bf16* __restrict__ Up,
                       const float*  __restrict__ Wb,
                       float*        __restrict__ h_out,
                       __bf16*       __restrict__ hbf,
                       float*        __restrict__ c,
                       int start, int m, int leaf)
{
    __shared__ __align__(16) unsigned char smem[16 * GATES * 4];   // 64 KB
    __bf16* ldsx  = (__bf16*)smem;                 // 16 x 320 bf16 (phase 1/2)
    __bf16* ldsc  = (__bf16*)(smem + 16*XKP*2);    // 16 x 512 bf16 (phase 1/2)
    float*  ldswx = (float*)smem;                  // 16 x 1024 f32 (phase 3/4)

    const int tid  = threadIdx.x;
    const int base = blockIdx.x * 16;              // node offset inside level
    const int cs   = 2 * start + 1;                // first child node index

    // ---- phase 1: stage A tiles into LDS (u32 copies of bf16 pairs) ----
    {
        const unsigned* xs = (const unsigned*)xbf;
        unsigned* xd = (unsigned*)ldsx;
        for (int i = tid; i < 16 * (XKP/2); i += 256) {
            int row = i / (XKP/2), kp = i - row * (XKP/2);
            int gn = base + row;
            unsigned v = 0u;
            if (gn < m) v = xs[(size_t)(start + gn) * (XKP/2) + kp];
            xd[row * (XKP/2) + kp] = v;
        }
        if (!leaf) {
            // ch_h row n = concat(h[cs+2gn], h[cs+2gn+1]) = 512 contiguous bf16
            const unsigned* hs = (const unsigned*)hbf;
            unsigned* hd = (unsigned*)ldsc;
            for (int i = tid; i < 16 * 256; i += 256) {
                int row = i >> 8, kp = i & 255;
                int gn = base + row;
                unsigned v = 0u;
                if (gn < m) v = hs[(size_t)(cs + 2 * gn) * (HS/2) + kp];
                hd[row * 256 + kp] = v;
            }
        }
    }
    __syncthreads();

    // ---- phase 2: WMMA GEMMs ----
    const int lane = tid & 31;
    const int wave = tid >> 5;
    const int lo   = lane & 15;       // M row of this lane
    const int hi   = lane >> 4;       // K-half selector (ISA 16-bit A layout)

    v8f acc[8];
#pragma unroll
    for (int t = 0; t < 8; ++t) acc[t] = (v8f){0,0,0,0,0,0,0,0};

    // W part: K = 320 (10 chunks of 32)
#pragma unroll
    for (int kc = 0; kc < KC_W; ++kc) {
        union { v16bf v; uint4 q[2]; } a;
        // A frag: lane lo holds row M=lo; elems 0..7 = K kc*32+hi*8..,
        //         elems 8..15 = K kc*32+16+hi*8.. (two ds_load_b128)
        a.q[0] = *(const uint4*)(ldsx + lo * XKP + kc * 32 + hi * 8);
        a.q[1] = *(const uint4*)(ldsx + lo * XKP + kc * 32 + 16 + hi * 8);
#pragma unroll
        for (int t = 0; t < 8; ++t) {
            int nt = wave * 8 + t;
            const v16bf b = *(const v16bf*)(Wp + (((size_t)kc * NT_TOT + nt) * 32 + lane) * 16);
            acc[t] = __builtin_amdgcn_wmma_f32_16x16x32_bf16(
                false, a.v, false, b, (short)0, acc[t], false, false);
        }
    }
    // U part: K = 512 (16 chunks of 32), children h
    if (!leaf) {
#pragma unroll
        for (int kc = 0; kc < KC_U; ++kc) {
            union { v16bf v; uint4 q[2]; } a;
            a.q[0] = *(const uint4*)(ldsc + lo * 512 + kc * 32 + hi * 8);
            a.q[1] = *(const uint4*)(ldsc + lo * 512 + kc * 32 + 16 + hi * 8);
#pragma unroll
            for (int t = 0; t < 8; ++t) {
                int nt = wave * 8 + t;
                const v16bf b = *(const v16bf*)(Up + (((size_t)kc * NT_TOT + nt) * 32 + lane) * 16);
                acc[t] = __builtin_amdgcn_wmma_f32_16x16x32_bf16(
                    false, a.v, false, b, (short)0, acc[t], false, false);
            }
        }
    }
    __syncthreads();   // everyone done reading staged A tiles

    // ---- phase 3: spill accumulators to LDS (C/D layout: VGPR g, lane L:
    //      M = g + 8*(L>=16), N = L%16) ----
#pragma unroll
    for (int t = 0; t < 8; ++t) {
        int ncol = (wave * 8 + t) * 16 + lo;
#pragma unroll
        for (int g = 0; g < 8; ++g)
            ldswx[(g + 8 * hi) * GATES + ncol] = acc[t][g];
    }
    __syncthreads();

    // ---- phase 4: gates (i,o,u[,f]) + cell/hidden updates ----
    for (int i2 = tid; i2 < 16 * HS; i2 += 256) {
        int n = i2 >> 8, j = i2 & 255;
        int gn = base + n;
        if (gn >= m) continue;
        float wi = ldswx[n * GATES +        j] + Wb[j];
        float wo = ldswx[n * GATES + HS   + j] + Wb[HS + j];
        float wu = ldswx[n * GATES + 2*HS + j] + Wb[2*HS + j];
        float ig = 1.0f / (1.0f + __expf(-wi));
        float og = 1.0f / (1.0f + __expf(-wo));
        float ug = tanhf(wu);
        float cl = ig * ug;
        if (!leaf) {
            float wf = ldswx[n * GATES + 3*HS + j] + Wb[3*HS + j];
            float fg = 1.0f / (1.0f + __expf(-wf));
            float csum = c[(size_t)(cs + 2*gn)     * HS + j]
                       + c[(size_t)(cs + 2*gn + 1) * HS + j];
            cl += fg * csum;
        }
        float hl = og * tanhf(cl);
        size_t node = (size_t)start + gn;
        c[node * HS + j]     = cl;
        h_out[node * HS + j] = hl;
        hbf[node * HS + j]   = (__bf16)hl;
    }
}

// ---------------------------------------------------------------------------
extern "C" void kernel_launch(void* const* d_in, const int* in_sizes, int n_in,
                              void* d_out, int out_size, void* d_ws, size_t ws_size,
                              hipStream_t stream)
{
    const float* x   = (const float*)d_in[0];   // (N, 300)
    const float* W_w = (const float*)d_in[1];   // (1024, 300)
    const float* W_b = (const float*)d_in[2];   // (1024,)
    const float* U_w = (const float*)d_in[3];   // (1024, 512)

    const long N = (long)in_sizes[0] / XK;      // 2^depth - 1
    int depth = 0;
    while (((1L << depth) - 1) < N) depth++;

    // workspace layout
    char* ws = (char*)d_ws;
    size_t o = 0;
    __bf16* Wp  = (__bf16*)(ws + o); o += (size_t)KC_W * NT_TOT * 32 * 16 * 2; // 640 KB
    __bf16* Up  = (__bf16*)(ws + o); o += (size_t)KC_U * NT_TOT * 32 * 16 * 2; // 1 MB
    __bf16* xbf = (__bf16*)(ws + o); o += (size_t)N * XKP * 2;                 // ~84 MB
    __bf16* hbf = (__bf16*)(ws + o); o += (size_t)N * HS * 2;                  // ~67 MB
    float*  c   = (float*)(ws + o);  o += (size_t)N * HS * 4;                  // ~134 MB

    // once-per-launch (deterministic) weight/x packing
    {
        int totW = KC_W * NT_TOT * 32 * 16;
        pack_b_kernel<<<(totW + 255) / 256, 256, 0, stream>>>(W_w, Wp, XK, totW);
        int totU = KC_U * NT_TOT * 32 * 16;
        pack_b_kernel<<<(totU + 255) / 256, 256, 0, stream>>>(U_w, Up, 512, totU);
        long totX = N * XKP;
        conv_x_kernel<<<(unsigned)((totX + 255) / 256), 256, 0, stream>>>(x, xbf, totX);
    }

    // bottom-up over levels; stream ordering provides the dependency chain
    for (int l = depth - 1; l >= 0; --l) {
        int m = 1 << l;
        int start = m - 1;
        int blocks = (m + 15) / 16;
        tree_level_kernel<<<blocks, 256, 0, stream>>>(
            xbf, Wp, Up, W_b, (float*)d_out, hbf, c, start, m, (l == depth - 1) ? 1 : 0);
    }
}
